// MultiHeadAttention_44684839748330
// MI455X (gfx1250) — matrix operand order repaired and checked by reading, not compiled, not run on hardware
//
#include <hip/hip_runtime.h>

// ---------------------------------------------------------------------------
// Multi-head attention forward for MI455X (gfx1250, wave32, WMMA bf16,
// async global->LDS staging with ASYNCcnt pipelining)
//   B=4, N=2048, DIM=1024, H=16, D=64
// ---------------------------------------------------------------------------

typedef __bf16 bf16_t;
typedef __attribute__((ext_vector_type(4)))  __bf16 v4bf;
typedef __attribute__((ext_vector_type(8)))  __bf16 v8bf;
typedef __attribute__((ext_vector_type(16))) __bf16 v16bf;
typedef __attribute__((ext_vector_type(4)))  float  v4f;
typedef __attribute__((ext_vector_type(8)))  float  v8f;

#define DIMC    1024
#define NHEADS  16
#define HDIM    64
#define BATCH   4
#define SEQ     2048
#define MROWS   (BATCH * SEQ)        // 8192
#define QK_SCALE 0.125f              // 1/sqrt(64)
#define LOG2E   1.44269504088896340736f
#define LDSP    40                   // LDS pitch (bf16) for 32-wide tiles: 80B
#define KPITCH  72                   // LDS pitch (bf16) for 64-wide K tiles: 144B

// ---------------------------------------------------------------------------
// WMMA + fragment helpers
// ---------------------------------------------------------------------------
__device__ __forceinline__ v16bf make_frag(v8bf a, v8bf b) {
    return __builtin_shufflevector(a, b, 0, 1, 2, 3, 4, 5, 6, 7,
                                          8, 9, 10, 11, 12, 13, 14, 15);
}

// A-operand (16x32, M on lanes): lane<16 holds K {0..7,16..23}, lane>=16 {8..15,24..31}
__device__ __forceinline__ v16bf load_frag_a(const bf16_t* row, int lane) {
    const int lo = (lane & 16) ? 8 : 0;
    v8bf x = *(const v8bf*)(row + lo);
    v8bf y = *(const v8bf*)(row + lo + 16);
    return make_frag(x, y);
}

// B-operand (32x16, N on lanes): lane<16 holds K {0..7,16..23}, lane>=16 {8..15,24..31}
__device__ __forceinline__ v16bf load_frag_b(const bf16_t* row, int lane) {
    const int base = (lane & 16) ? 8 : 0;
    v8bf x = *(const v8bf*)(row + base);
    v8bf y = *(const v8bf*)(row + base + 16);
    return make_frag(x, y);
}

__device__ __forceinline__ v8f wmma_bf16(v16bf a, v16bf b, v8f c) {
    return __builtin_amdgcn_wmma_f32_16x16x32_bf16(
        false, a, false, b, (short)0, c, false, false);
}

// ---------------------------------------------------------------------------
// Async global->LDS copy (CDNA5 GLOBAL_LOAD_ASYNC_TO_LDS, GVS mode).
// Tracked with ASYNCcnt; waited with s_wait_asynccnt.
// ---------------------------------------------------------------------------
__device__ __forceinline__ unsigned lds_off(const void* p) {
    // low 32 bits of a generic shared-aperture address == LDS byte offset
    return (unsigned)(uintptr_t)p;
}

__device__ __forceinline__ void async_b128(unsigned lds_byte, unsigned goff_byte,
                                           const bf16_t* base) {
    asm volatile("global_load_async_to_lds_b128 %0, %1, %2"
                 :: "v"(lds_byte), "v"(goff_byte), "s"(base)
                 : "memory");
}

__device__ __forceinline__ void wait_async0() {
    asm volatile("s_wait_asynccnt 0x0" ::: "memory");
}

// ---------------------------------------------------------------------------
// Kernel 0: fp32 -> bf16 weight conversion
// ---------------------------------------------------------------------------
__global__ __launch_bounds__(256)
void cvt_f32_bf16_kernel(const float* __restrict__ src, bf16_t* __restrict__ dst, int n) {
    int i = (blockIdx.x * 256 + threadIdx.x) * 4;
    if (i >= n) return;
    v4f f = *(const v4f*)(src + i);
    v4bf h;
    h[0] = (bf16_t)f[0]; h[1] = (bf16_t)f[1];
    h[2] = (bf16_t)f[2]; h[3] = (bf16_t)f[3];
    *(v4bf*)(dst + i) = h;
}

// ---------------------------------------------------------------------------
// Kernel 1: projection GEMM  C[M,N] = X[M,K] * W[N,K]^T + bias
//   X fp32 (converted in registers), W bf16 (async-staged), out bf16.
//   mode 0: out[b,h,tok,d]   mode 1: out[b,h,d,tok] (V transposed)
// ---------------------------------------------------------------------------
__global__ __launch_bounds__(256)
void proj_gemm_kernel(const float* __restrict__ X, const bf16_t* __restrict__ W,
                      const float* __restrict__ bias, bf16_t* __restrict__ out,
                      int mode, float oscale) {
    __shared__ __align__(16) bf16_t Asl[128 * LDSP];
    __shared__ __align__(16) bf16_t Bsl[64 * LDSP];

    const int tid = threadIdx.x;
    const int lane = tid & 31;
    const int wave = tid >> 5;
    const int wm = wave >> 1, wn = wave & 1;
    const int lr = lane & 15;
    const int m0 = blockIdx.x * 128;
    const int n0 = blockIdx.y * 64;
    const unsigned bslBase = lds_off(Bsl);

    v8f acc[2][2];
#pragma unroll
    for (int i = 0; i < 2; i++)
#pragma unroll
        for (int j = 0; j < 2; j++) acc[i][j] = (v8f)(0.0f);

    for (int k0 = 0; k0 < DIMC; k0 += 32) {
        __syncthreads();   // previous iteration's LDS reads complete
        {   // async-stage B tile: 64x32 bf16, 16B per thread
            const int row = tid >> 2;
            const int kq = (tid & 3) * 8;
            async_b128(bslBase + (unsigned)(row * LDSP + kq) * 2,
                       (unsigned)(((n0 + row) * DIMC + k0 + kq) * 2), W);
        }
        {   // stage A tile: 128x32 fp32 -> bf16 via registers
            const int row = tid >> 1;
            const int kq = (tid & 1) * 16;
            const float* src = X + (size_t)(m0 + row) * DIMC + (k0 + kq);
            if (k0 + 32 < DIMC) __builtin_prefetch(src + 32, 0, 1);
            v4f f0 = *(const v4f*)(src + 0);
            v4f f1 = *(const v4f*)(src + 4);
            v4f f2 = *(const v4f*)(src + 8);
            v4f f3 = *(const v4f*)(src + 12);
            v8bf h0, h1;
#pragma unroll
            for (int j = 0; j < 4; j++) {
                h0[j] = (bf16_t)f0[j]; h0[4 + j] = (bf16_t)f1[j];
                h1[j] = (bf16_t)f2[j]; h1[4 + j] = (bf16_t)f3[j];
            }
            *(v8bf*)(Asl + row * LDSP + kq) = h0;
            *(v8bf*)(Asl + row * LDSP + kq + 8) = h1;
        }
        wait_async0();
        __syncthreads();

        v16bf af[2], bfr[2];
#pragma unroll
        for (int tm = 0; tm < 2; tm++)
            af[tm] = load_frag_a(Asl + (wm * 32 + tm * 16 + lr) * LDSP, lane);
#pragma unroll
        for (int tn = 0; tn < 2; tn++)
            bfr[tn] = load_frag_b(Bsl + (wn * 32 + tn * 16 + lr) * LDSP, lane);
#pragma unroll
        for (int tm = 0; tm < 2; tm++)
#pragma unroll
            for (int tn = 0; tn < 2; tn++)
                acc[tm][tn] = wmma_bf16(af[tm], bfr[tn], acc[tm][tn]);
    }

#pragma unroll
    for (int tm = 0; tm < 2; tm++) {
        const int mbase = m0 + wm * 32 + tm * 16 + ((lane & 16) ? 8 : 0);
#pragma unroll
        for (int tn = 0; tn < 2; tn++) {
            const int gn = n0 + wn * 32 + tn * 16 + lr;
            const float bv = bias[gn];
            const int h = gn >> 6, d = gn & 63;
#pragma unroll
            for (int r = 0; r < 8; r++) {
                const int gm = mbase + r;
                const int b = gm >> 11, tok = gm & 2047;
                const float v = (acc[tm][tn][r] + bv) * oscale;
                size_t off;
                if (mode == 0)
                    off = (((size_t)(b * NHEADS + h)) * SEQ + tok) * HDIM + d;
                else
                    off = (((size_t)(b * NHEADS + h)) * HDIM + d) * SEQ + tok;
                out[off] = (bf16_t)v;
            }
        }
    }
}

// ---------------------------------------------------------------------------
// Kernel 2: flash attention with double-buffered async K/V staging.
//   Qh, Kh: [B,H,N,D] bf16 (Q pre-scaled);  Vt: [B,H,D,N] bf16
//   ctx: [B*N, DIM] bf16
// Block: 256 thr = 8 waves; wave owns 16 q rows; block streams K/V chunks of 32
// tokens through LDS (async copy overlapped with WMMA via ASYNCcnt).
// ---------------------------------------------------------------------------
__device__ __forceinline__ void stage_kv_async(int tid, int kc,
                                               bf16_t* kdst, bf16_t* vdst,
                                               const bf16_t* Kb, const bf16_t* Vb) {
    {   // K tile: 32 tokens x 64 d  (row-major, pitch KPITCH)
        const int r = tid >> 3;           // 0..31 token row
        const int c = (tid & 7) * 8;      // d chunk
        async_b128(lds_off(kdst + r * KPITCH + c),
                   (unsigned)(((kc + r) * HDIM + c) * 2), Kb);
    }
    {   // V tile: 64 d-rows x 32 tokens (from Vt, pitch LDSP)
        const int r = tid >> 2;           // 0..63 d row
        const int c = (tid & 3) * 8;      // token chunk
        async_b128(lds_off(vdst + r * LDSP + c),
                   (unsigned)((r * SEQ + kc + c) * 2), Vb);
    }
}

__global__ __launch_bounds__(256)
void attn_kernel(const bf16_t* __restrict__ Qh, const bf16_t* __restrict__ Kh,
                 const bf16_t* __restrict__ Vt, bf16_t* __restrict__ ctx) {
    __shared__ __align__(16) bf16_t Ksl[2][32 * KPITCH];
    __shared__ __align__(16) bf16_t Vsl[2][64 * LDSP];
    __shared__ __align__(16) bf16_t Pls[8][16 * LDSP];

    const int tid = threadIdx.x;
    const int lane = tid & 31;
    const int wave = tid >> 5;
    const int lr = lane & 15;
    const int hi8 = (lane & 16) ? 8 : 0;

    const int bh = blockIdx.x;                       // b*16 + h
    const int q0 = blockIdx.y * 128 + wave * 16;

    const bf16_t* Qb = Qh + (size_t)bh * SEQ * HDIM;
    const bf16_t* Kb = Kh + (size_t)bh * SEQ * HDIM;
    const bf16_t* Vb = Vt + (size_t)bh * HDIM * SEQ;
    bf16_t* Pw = &Pls[wave][0];

    // resident Q fragments (A-operand), d-halves 0..31 / 32..63
    v16bf qf[2];
#pragma unroll
    for (int dh = 0; dh < 2; dh++)
        qf[dh] = load_frag_a(Qb + (size_t)(q0 + lr) * HDIM + dh * 32, lane);

    v8f o[4];
#pragma unroll
    for (int t = 0; t < 4; t++) o[t] = (v8f)(0.0f);
    float mstat[8], lstat[8];
#pragma unroll
    for (int r = 0; r < 8; r++) { mstat[r] = -1e30f; lstat[r] = 0.0f; }

    int buf = 0;
    stage_kv_async(tid, 0, Ksl[0], Vsl[0], Kb, Vb);

    for (int kc = 0; kc < SEQ; kc += 32) {
        wait_async0();       // this wave's chunk-copy landed in LDS
        __syncthreads();     // all waves' copies landed; prev buffer reads done
        if (kc + 32 < SEQ)   // overlap next chunk's copy with this chunk's math
            stage_kv_async(tid, kc + 32, Ksl[buf ^ 1], Vsl[buf ^ 1], Kb, Vb);

        const bf16_t* Kt = Ksl[buf];
        const bf16_t* Vtile = Vsl[buf];

        // ---- S = Q * K^T for 32 key columns (two 16x16 tiles) ----
        v8f s[2];
#pragma unroll
        for (int t = 0; t < 2; t++) {
            s[t] = (v8f)(0.0f);
#pragma unroll
            for (int dh = 0; dh < 2; dh++) {
                v16bf kf = load_frag_b(Kt + (t * 16 + lr) * KPITCH + dh * 32, lane);
                s[t] = wmma_bf16(qf[dh], kf, s[t]);
            }
        }

        // ---- online softmax update ----
        float alpha[8];
#pragma unroll
        for (int r = 0; r < 8; r++) {
            float mx = fmaxf(s[0][r], s[1][r]);
#pragma unroll
            for (int msk = 8; msk >= 1; msk >>= 1)
                mx = fmaxf(mx, __shfl_xor(mx, msk, 32));
            const float mnew = fmaxf(mstat[r], mx);
            alpha[r] = __builtin_amdgcn_exp2f((mstat[r] - mnew) * LOG2E);
            mstat[r] = mnew;

            const float p0 = __builtin_amdgcn_exp2f((s[0][r] - mnew) * LOG2E);
            const float p1 = __builtin_amdgcn_exp2f((s[1][r] - mnew) * LOG2E);
            float rs = p0 + p1;
#pragma unroll
            for (int msk = 8; msk >= 1; msk >>= 1)
                rs += __shfl_xor(rs, msk, 32);
            lstat[r] = lstat[r] * alpha[r] + rs;

            const int rr = r + hi8;                 // C-layout row
            Pw[rr * LDSP + lr] = (bf16_t)p0;
            Pw[rr * LDSP + 16 + lr] = (bf16_t)p1;
        }
        // per-wave LDS RAW on P-transpose buffer
        asm volatile("s_wait_dscnt 0x0" ::: "memory");
        v16bf pf = load_frag_a(Pw + lr * LDSP, lane);

        // ---- O = O*alpha + P * V ----
#pragma unroll
        for (int td = 0; td < 4; td++) {
#pragma unroll
            for (int r = 0; r < 8; r++) o[td][r] *= alpha[r];
            v16bf vf = load_frag_b(Vtile + (td * 16 + lr) * LDSP, lane);
            o[td] = wmma_bf16(pf, vf, o[td]);
        }
        buf ^= 1;
    }

    // ---- finalize: divide by l, merge heads into ctx[B*N, DIM] ----
    float inv[8];
#pragma unroll
    for (int r = 0; r < 8; r++) inv[r] = 1.0f / lstat[r];
    const int b = bh >> 4, h = bh & 15;
#pragma unroll
    for (int td = 0; td < 4; td++) {
        const int d = td * 16 + lr;
#pragma unroll
        for (int r = 0; r < 8; r++) {
            const int q = q0 + r + hi8;
            ctx[((size_t)(b * SEQ + q)) * DIMC + h * HDIM + d] =
                (bf16_t)(o[td][r] * inv[r]);
        }
    }
}

// ---------------------------------------------------------------------------
// Kernel 3: output projection  out[M,N] = ctx[M,K] * Wo[N,K]^T + bo (fp32 out)
// Both tiles are bf16 copies -> fully async-staged.
// ---------------------------------------------------------------------------
__global__ __launch_bounds__(256)
void oproj_gemm_kernel(const bf16_t* __restrict__ A, const bf16_t* __restrict__ W,
                       const float* __restrict__ bias, float* __restrict__ out) {
    __shared__ __align__(16) bf16_t Asl[128 * LDSP];
    __shared__ __align__(16) bf16_t Bsl[64 * LDSP];

    const int tid = threadIdx.x;
    const int lane = tid & 31;
    const int wave = tid >> 5;
    const int wm = wave >> 1, wn = wave & 1;
    const int lr = lane & 15;
    const int m0 = blockIdx.x * 128;
    const int n0 = blockIdx.y * 64;
    const unsigned aslBase = lds_off(Asl);
    const unsigned bslBase = lds_off(Bsl);

    v8f acc[2][2];
#pragma unroll
    for (int i = 0; i < 2; i++)
#pragma unroll
        for (int j = 0; j < 2; j++) acc[i][j] = (v8f)(0.0f);

    for (int k0 = 0; k0 < DIMC; k0 += 32) {
        __syncthreads();
        {   // async A tile: 128x32 bf16, 32B per thread (two b128)
            const int row = tid >> 1;
            const int kq = (tid & 1) * 16;
            const unsigned l = aslBase + (unsigned)(row * LDSP + kq) * 2;
            const unsigned g = (unsigned)(((m0 + row) * DIMC + k0 + kq) * 2);
            async_b128(l, g, A);
            async_b128(l + 16, g + 16, A);
        }
        {   // async B tile: 64x32 bf16, 16B per thread
            const int row = tid >> 2;
            const int kq = (tid & 3) * 8;
            async_b128(bslBase + (unsigned)(row * LDSP + kq) * 2,
                       (unsigned)(((n0 + row) * DIMC + k0 + kq) * 2), W);
        }
        wait_async0();
        __syncthreads();

        v16bf af[2], bfr[2];
#pragma unroll
        for (int tm = 0; tm < 2; tm++)
            af[tm] = load_frag_a(Asl + (wm * 32 + tm * 16 + lr) * LDSP, lane);
#pragma unroll
        for (int tn = 0; tn < 2; tn++)
            bfr[tn] = load_frag_b(Bsl + (wn * 32 + tn * 16 + lr) * LDSP, lane);
#pragma unroll
        for (int tm = 0; tm < 2; tm++)
#pragma unroll
            for (int tn = 0; tn < 2; tn++)
                acc[tm][tn] = wmma_bf16(af[tm], bfr[tn], acc[tm][tn]);
    }

#pragma unroll
    for (int tm = 0; tm < 2; tm++) {
        const int mbase = m0 + wm * 32 + tm * 16 + ((lane & 16) ? 8 : 0);
#pragma unroll
        for (int tn = 0; tn < 2; tn++) {
            const int gn = n0 + wn * 32 + tn * 16 + lr;
            const float bv = bias[gn];
#pragma unroll
            for (int r = 0; r < 8; r++)
                out[(size_t)(mbase + r) * DIMC + gn] = acc[tm][tn][r] + bv;
        }
    }
}

// ---------------------------------------------------------------------------
// Host launcher
// ---------------------------------------------------------------------------
extern "C" void kernel_launch(void* const* d_in, const int* in_sizes, int n_in,
                              void* d_out, int out_size, void* d_ws, size_t ws_size,
                              hipStream_t stream) {
    (void)in_sizes; (void)n_in; (void)out_size; (void)ws_size;
    const float* q  = (const float*)d_in[0];
    const float* k  = (const float*)d_in[1];
    const float* v  = (const float*)d_in[2];
    const float* wq = (const float*)d_in[3];
    const float* bq = (const float*)d_in[4];
    const float* wk = (const float*)d_in[5];
    const float* bk = (const float*)d_in[6];
    const float* wv = (const float*)d_in[7];
    const float* bv = (const float*)d_in[8];
    const float* wo = (const float*)d_in[9];
    const float* bo = (const float*)d_in[10];
    float* out = (float*)d_out;

    const size_t WW = (size_t)DIMC * DIMC;           // 1,048,576
    const size_t HT = (size_t)BATCH * SEQ * DIMC;    // 8,388,608
    bf16_t* wsb = (bf16_t*)d_ws;
    bf16_t* wqb = wsb;
    bf16_t* wkb = wsb + WW;
    bf16_t* wvb = wsb + 2 * WW;
    bf16_t* wob = wsb + 3 * WW;
    bf16_t* Qh  = wsb + 4 * WW;
    bf16_t* Kh  = Qh + HT;
    bf16_t* Vt  = Kh + HT;
    bf16_t* ctx = Vt + HT;

    // 0) weights fp32 -> bf16
    const int cvtBlocks = (int)(WW / (256 * 4));
    cvt_f32_bf16_kernel<<<cvtBlocks, 256, 0, stream>>>(wq, wqb, (int)WW);
    cvt_f32_bf16_kernel<<<cvtBlocks, 256, 0, stream>>>(wk, wkb, (int)WW);
    cvt_f32_bf16_kernel<<<cvtBlocks, 256, 0, stream>>>(wv, wvb, (int)WW);
    cvt_f32_bf16_kernel<<<cvtBlocks, 256, 0, stream>>>(wo, wob, (int)WW);

    // 1) fused projections (Q scaled by 1/sqrt(D); V stored transposed)
    dim3 ggrid(MROWS / 128, DIMC / 64);
    proj_gemm_kernel<<<ggrid, 256, 0, stream>>>(q, wqb, bq, Qh, 0, QK_SCALE);
    proj_gemm_kernel<<<ggrid, 256, 0, stream>>>(k, wkb, bk, Kh, 0, 1.0f);
    proj_gemm_kernel<<<ggrid, 256, 0, stream>>>(v, wvb, bv, Vt, 1, 1.0f);

    // 2) flash attention (double-buffered async K/V staging)
    attn_kernel<<<dim3(BATCH * NHEADS, SEQ / 128), 256, 0, stream>>>(Qh, Kh, Vt, ctx);

    // 3) output projection -> fp32
    oproj_gemm_kernel<<<ggrid, 256, 0, stream>>>(ctx, wob, bo, out);
}
